// ConvGRUDecoder_25099788878594
// MI455X (gfx1250) — compile-verified, matrix operand order checked by reading, not run on hardware
//
#include <hip/hip_runtime.h>
#include <hip/hip_bf16.h>

typedef _Float16 v16h __attribute__((ext_vector_type(16)));
typedef _Float16 v8h  __attribute__((ext_vector_type(8)));
typedef float    v8f  __attribute__((ext_vector_type(8)));

#define HD    128
#define KTOT  192
#define HW    (512*512)

// ---- LDS layout (bytes) ----
#define SZ_W     (HD*KTOT*2)          // 49152 per gate weight (f16)
#define OFF_WZ   0
#define OFF_WR   (OFF_WZ + SZ_W)
#define OFF_WQ   (OFF_WR + SZ_W)
#define OFF_W1   (OFF_WQ + SZ_W)      // 147456
#define SZ_W1    (32*KTOT*2)          // 12288
#define OFF_BZ   (OFF_W1 + SZ_W1)     // 159744
#define OFF_BR   (OFF_BZ + 512)
#define OFF_BQ   (OFF_BR + 512)
#define OFF_B1   (OFF_BQ + 512)       // 161280
#define OFF_W2   (OFF_B1 + 128)       // 161408 (3*32 f32)
#define OFF_B2   (OFF_W2 + 384)       // 161792
#define OFF_HX   (OFF_B2 + 32)        // 161824, 16B aligned
#define SZ_HX    (16*KTOT*2)          // 6144 per wave
#define OFF_HID  (OFF_HX + 8*SZ_HX)   // 210976
#define SZ_HID   (16*32*4)            // 2048 per wave
#define OFF_ZB   (OFF_HID + 8*SZ_HID) // 227360
#define SZ_ZB    (16*HD*2)            // 4096 per wave (z gate, f16)
#define SMEM_TOTAL (OFF_ZB + 8*SZ_ZB) // 260128 bytes < 320KB WGP LDS

static __device__ __forceinline__ v8f vzero8() {
    v8f z = {0.f,0.f,0.f,0.f,0.f,0.f,0.f,0.f};
    return z;
}

// Load a 16-half WMMA fragment: halves [0..7] at p, [8..15] at p+16.
static __device__ __forceinline__ v16h ld_frag(const _Float16* p) {
    v8h lo = *(const v8h*)(p);
    v8h hi = *(const v8h*)(p + 16);
    return __builtin_shufflevector(lo, hi, 0,1,2,3,4,5,6,7,8,9,10,11,12,13,14,15);
}

// gfx1250 native v_tanh_f32 (TRANS pipe), with fallbacks.
#if __has_builtin(__builtin_amdgcn_tanhf)
#define FAST_TANH(x) __builtin_amdgcn_tanhf(x)
#elif __has_builtin(__builtin_amdgcn_tanh_f32)
#define FAST_TANH(x) __builtin_amdgcn_tanh_f32(x)
#else
#define FAST_TANH(x) tanhf(x)
#endif

static __device__ __forceinline__ float geluf_(float x) {
    return 0.5f * x * (1.0f + erff(x * 0.70710678118654752f));
}

#define WMMA_F16(A,B,C) \
    __builtin_amdgcn_wmma_f32_16x16x32_f16(false, (A), false, (B), (short)0, (C), false, false)

#define DS_FENCE() asm volatile("s_wait_dscnt 0" ::: "memory")

// 16x128x192 gate GEMM with software-pipelined LDS fragment fetch:
//   depth-2 prefetch on B fragments, depth-1 on A fragments.
// bB/aB are lane-adjusted base pointers (already include lhalf*KTOT + khi).
static __device__ __forceinline__ void gate_gemm(const _Float16* __restrict__ bB,
                                                 const _Float16* __restrict__ aB,
                                                 v8f acc[8]) {
    v16h a  = ld_frag(aB);
    v16h b0 = ld_frag(bB);
    v16h b1 = ld_frag(bB + 16*KTOT);
    v16h aN = a;
    #pragma unroll
    for (int i = 0; i < 48; ++i) {
        const int kc = i >> 3, nt = i & 7;
        v16h bp = b1;
        if (i + 2 < 48) {
            const int j = i + 2;
            bp = ld_frag(bB + (j & 7)*16*KTOT + (j >> 3)*32);
        }
        if (nt == 0 && kc < 5) aN = ld_frag(aB + (kc + 1)*32);
        if (kc == 0) acc[nt] = WMMA_F16(a, b0, vzero8());   // C = inline 0
        else         acc[nt] = WMMA_F16(a, b0, acc[nt]);
        b0 = b1; b1 = bp;
        if (nt == 7) a = aN;
    }
}

__global__ __launch_bounds__(256, 1)
void convgru_decoder_kernel(const float* __restrict__ beforeI,
                            const float* __restrict__ afterI,
                            const float* __restrict__ po,
                            const int*   __restrict__ vc,
                            const float* __restrict__ Wo, const float* __restrict__ bo,
                            const float* __restrict__ Wz, const float* __restrict__ bz,
                            const float* __restrict__ Wr, const float* __restrict__ br,
                            const float* __restrict__ Wq, const float* __restrict__ bq,
                            const float* __restrict__ W1, const float* __restrict__ b1,
                            const float* __restrict__ W2, const float* __restrict__ b2,
                            float* __restrict__ out, int N, int numTiles)
{
    extern __shared__ char smem[];
    _Float16* ldsWZ = (_Float16*)(smem + OFF_WZ);
    _Float16* ldsWR = (_Float16*)(smem + OFF_WR);
    _Float16* ldsWQ = (_Float16*)(smem + OFF_WQ);
    _Float16* ldsW1 = (_Float16*)(smem + OFF_W1);
    float*    ldsBZ = (float*)(smem + OFF_BZ);
    float*    ldsBR = (float*)(smem + OFF_BR);
    float*    ldsBQ = (float*)(smem + OFF_BQ);
    float*    ldsB1 = (float*)(smem + OFF_B1);
    float*    ldsW2 = (float*)(smem + OFF_W2);
    float*    ldsB2 = (float*)(smem + OFF_B2);

    const int tid = threadIdx.x;

    // ---- cooperative weight preload: f32 global -> f16 LDS ----
    for (int i = tid; i < HD*KTOT; i += 256) {
        ldsWZ[i] = (_Float16)Wz[i];
        ldsWR[i] = (_Float16)Wr[i];
        ldsWQ[i] = (_Float16)Wq[i];
    }
    for (int i = tid; i < 32*KTOT; i += 256) ldsW1[i] = (_Float16)W1[i];
    for (int i = tid; i < HD; i += 256) {
        ldsBZ[i] = bz[i]; ldsBR[i] = br[i]; ldsBQ[i] = bq[i];
    }
    if (tid < 32) ldsB1[tid] = b1[tid];
    if (tid < 96) ldsW2[tid] = W2[tid];
    if (tid < 3)  ldsB2[tid] = b2[tid];
    __syncthreads();

    const int wid   = tid >> 5;
    const int lane  = tid & 31;
    const int lhalf = lane & 15;         // column / M index within fragment
    const int khi   = (lane >> 4) * 8;   // +8 half offset for upper half-wave
    const int rowhi = (lane >> 4) * 8;   // D-layout: lanes 16-31 hold rows 8..15

    _Float16* hx   = (_Float16*)(smem + OFF_HX  + wid * SZ_HX);   // [16][192] f16
    float*    hidL = (float*)   (smem + OFF_HID + wid * SZ_HID);  // [16][32]  f32
    _Float16* zbuf = (_Float16*)(smem + OFF_ZB  + wid * SZ_ZB);   // [16][128] f16

    // lane-adjusted fragment base pointers
    const _Float16* aB  = hx    + lhalf*KTOT + khi;
    const _Float16* bBZ = ldsWZ + lhalf*KTOT + khi;
    const _Float16* bBR = ldsWR + lhalf*KTOT + khi;
    const _Float16* bBQ = ldsWQ + lhalf*KTOT + khi;
    const _Float16* bB1 = ldsW1 + lhalf*KTOT + khi;

    // per-lane-constant input projection coefficients
    float woA[4], woB[4], woC[4], boL[4];
    #pragma unroll
    for (int xt = 0; xt < 4; ++xt) {
        const int ch = xt * 16 + lhalf;
        woA[xt] = Wo[ch*3+0];
        woB[xt] = Wo[ch*3+1];
        woC[xt] = Wo[ch*3+2];
        boL[xt] = bo[ch];
    }

    for (int t = blockIdx.x * 8 + wid; t < numTiles; t += gridDim.x * 8) {
        const int p0 = t * 16;

        // ---- init: gather h = [before, after], compute x = po@Wo.T + bo ----
        v8f h[8];
        #pragma unroll
        for (int v = 0; v < 8; ++v) {
            const int mloc = v + rowhi;
            int m = p0 + mloc; m = (m < N) ? m : (N - 1);
            const int yy  = vc[m*3 + 1];
            const int xx  = vc[m*3 + 2];
            const int pix = yy * 512 + xx;
            #pragma unroll
            for (int nt = 0; nt < 8; ++nt) {
                const float* img = (nt < 4) ? beforeI : afterI;
                const int ch = (nt & 3) * 16 + lhalf;
                h[nt][v] = img[ch * HW + pix];
            }
            const float pa = po[m*3+0], pb = po[m*3+1], pc = po[m*3+2];
            #pragma unroll
            for (int xt = 0; xt < 4; ++xt) {
                float xv = fmaf(pa, woA[xt], fmaf(pb, woB[xt], fmaf(pc, woC[xt], boL[xt])));
                hx[mloc * KTOT + HD + xt*16 + lhalf] = (_Float16)xv;
            }
            #pragma unroll
            for (int nt = 0; nt < 8; ++nt)
                hx[mloc * KTOT + nt*16 + lhalf] = (_Float16)h[nt][v];
        }
        DS_FENCE();

        // ---- GRU iterations (one 64-VGPR accumulator array live at a time) ----
        #pragma unroll 1
        for (int it = 0; it < 4; ++it) {
            v8f acc[8];

            // ---- z gate: sigmoid(hx @ Wz.T + bz) -> zbuf (f16) ----
            gate_gemm(bBZ, aB, acc);
            #pragma unroll
            for (int nt = 0; nt < 8; ++nt) {
                const float bh = 0.5f * ldsBZ[nt*16 + lhalf];
                #pragma unroll
                for (int v = 0; v < 8; ++v) {
                    // sigmoid(x+b) = 0.5 + 0.5*tanh(0.5*x + 0.5*b)
                    const float zz = fmaf(FAST_TANH(fmaf(acc[nt][v], 0.5f, bh)), 0.5f, 0.5f);
                    zbuf[(v + rowhi)*HD + nt*16 + lhalf] = (_Float16)zz;
                }
            }

            // ---- r gate: sigmoid(hx @ Wr.T + br); r*h -> hx h-part (f16) ----
            gate_gemm(bBR, aB, acc);
            #pragma unroll
            for (int nt = 0; nt < 8; ++nt) {
                const float bh = 0.5f * ldsBR[nt*16 + lhalf];
                #pragma unroll
                for (int v = 0; v < 8; ++v) {
                    const float rr = fmaf(FAST_TANH(fmaf(acc[nt][v], 0.5f, bh)), 0.5f, 0.5f);
                    hx[(v + rowhi)*KTOT + nt*16 + lhalf] = (_Float16)(rr * h[nt][v]);
                }
            }
            DS_FENCE();

            // ---- q gate: tanh([r*h, x] @ Wq.T + bq); h update ----
            gate_gemm(bBQ, aB, acc);
            #pragma unroll
            for (int nt = 0; nt < 8; ++nt) {
                const float bqv = ldsBQ[nt*16 + lhalf];
                #pragma unroll
                for (int v = 0; v < 8; ++v) {
                    const float qq = FAST_TANH(acc[nt][v] + bqv);
                    const float zz = (float)zbuf[(v + rowhi)*HD + nt*16 + lhalf];
                    const float hv = h[nt][v];
                    const float hn = fmaf(zz, qq - hv, hv);   // (1-z)h + zq
                    h[nt][v] = hn;
                    hx[(v + rowhi)*KTOT + nt*16 + lhalf] = (_Float16)hn;
                }
            }
            DS_FENCE();
        }

        // ---- epilogue: hid = gelu(feat @ W1.T + b1) ----
        v8f f0, f1;
        {
            v16h a0  = ld_frag(aB);
            v16h b0  = ld_frag(bB1);
            v16h b1f = ld_frag(bB1 + 16*KTOT);
            f0 = WMMA_F16(a0, b0, vzero8());
            f1 = WMMA_F16(a0, b1f, vzero8());
        }
        #pragma unroll
        for (int kc = 1; kc < 6; ++kc) {
            v16h a   = ld_frag(aB + kc*32);
            v16h b0  = ld_frag(bB1 + kc*32);
            v16h b1f = ld_frag(bB1 + 16*KTOT + kc*32);
            f0 = WMMA_F16(a, b0, f0);
            f1 = WMMA_F16(a, b1f, f1);
        }
        const float b1a = ldsB1[lhalf], b1b = ldsB1[16 + lhalf];
        #pragma unroll
        for (int v = 0; v < 8; ++v) {
            hidL[(v + rowhi)*32 + lhalf]      = geluf_(f0[v] + b1a);
            hidL[(v + rowhi)*32 + 16 + lhalf] = geluf_(f1[v] + b1b);
        }
        DS_FENCE();

        // ---- flow = hid @ W2.T + b2 (N=3: per-lane dot products) ----
        if (lane < 16) {
            const int m = p0 + lane;
            if (m < N) {
                float o0 = ldsB2[0], o1 = ldsB2[1], o2 = ldsB2[2];
                #pragma unroll
                for (int n = 0; n < 32; ++n) {
                    const float hv = hidL[lane*32 + n];
                    o0 = fmaf(hv, ldsW2[0*32 + n], o0);
                    o1 = fmaf(hv, ldsW2[1*32 + n], o1);
                    o2 = fmaf(hv, ldsW2[2*32 + n], o2);
                }
                out[m*3 + 0] = o0;
                out[m*3 + 1] = o1;
                out[m*3 + 2] = o2;
            }
        }
    }
}

extern "C" void kernel_launch(void* const* d_in, const int* in_sizes, int n_in,
                              void* d_out, int out_size, void* d_ws, size_t ws_size,
                              hipStream_t stream) {
    const float* beforeI = (const float*)d_in[0];
    const float* afterI  = (const float*)d_in[1];
    const float* po      = (const float*)d_in[2];
    const int*   vc      = (const int*)  d_in[3];
    const float* Wo      = (const float*)d_in[4];
    const float* bo      = (const float*)d_in[5];
    const float* Wz      = (const float*)d_in[6];
    const float* bz      = (const float*)d_in[7];
    const float* Wr      = (const float*)d_in[8];
    const float* br      = (const float*)d_in[9];
    const float* Wq      = (const float*)d_in[10];
    const float* bq      = (const float*)d_in[11];
    const float* W1      = (const float*)d_in[12];
    const float* b1      = (const float*)d_in[13];
    const float* W2      = (const float*)d_in[14];
    const float* b2      = (const float*)d_in[15];
    float* out = (float*)d_out;

    const int N = in_sizes[2] / 3;
    const int numTiles = (N + 15) / 16;
    int blocks = (numTiles + 7) / 8;
    if (blocks > 1024) blocks = 1024;

    (void)hipFuncSetAttribute((const void*)convgru_decoder_kernel,
                              hipFuncAttributeMaxDynamicSharedMemorySize,
                              SMEM_TOTAL);

    convgru_decoder_kernel<<<blocks, 256, SMEM_TOTAL, stream>>>(
        beforeI, afterI, po, vc, Wo, bo, Wz, bz, Wr, br, Wq, bq,
        W1, b1, W2, b2, out, N, numTiles);
}
